// GAT_Encoder_19430432047705
// MI455X (gfx1250) — compile-verified, matrix operand order checked by reading, not run on hardware
//
#include <hip/hip_runtime.h>
#include <hip/hip_bf16.h>

// ---------------- problem constants ----------------
constexpr int cB  = 16;
constexpr int cN  = 256;
constexpr int cF  = 1024;
constexpr int cH  = 16;
constexpr int cHD = 64;          // head dim
constexpr float NEGV = -9e15f;

// ---------------- WMMA types ----------------
typedef __attribute__((ext_vector_type(16))) __bf16 v16bf;
typedef __attribute__((ext_vector_type(8)))  float  v8f;
typedef __attribute__((ext_vector_type(8)))  unsigned int v8u;

union Frag16 { v8u u; v16bf b; };

__device__ __forceinline__ unsigned short f2bf(float x) {
    // round-to-nearest-even fp32 -> bf16
    unsigned int u = __float_as_uint(x);
    unsigned int r = u + 0x7FFFu + ((u >> 16) & 1u);
    return (unsigned short)(r >> 16);
}
__device__ __forceinline__ unsigned int pack2bf(float a, float b) {
    return (unsigned int)f2bf(a) | ((unsigned int)f2bf(b) << 16);
}

// k-sub-index inside a 16x32 (or 32x16) bf16 WMMA fragment, per ISA 7.12.2:
// vgpr r, lane-group g: kk = (r&3)*2 + (r>>2)*16 + g*8  (pairs {kk,kk+1} consecutive)
__device__ __forceinline__ int frag_kpair(int r, int g) {
    return ((r & 3) << 1) + ((r >> 2) << 4) + (g << 3);
}

// Load a 16x32 A-fragment (or 32x16 B-fragment of C = A @ W^T) from a
// row-major bf16 matrix: element(row0 + lane%16, koff + kk).
// Contiguous kk/2 for r=0..3 and r=4..7 -> compiler emits 2x global_load_b128.
__device__ __forceinline__ v16bf load_frag_g(const unsigned short* base,
                                             int row0, int ld, int koff, int lane) {
    const int g = lane >> 4, idx = lane & 15;
    const unsigned int* p =
        reinterpret_cast<const unsigned int*>(base + (size_t)(row0 + idx) * ld + koff);
    Frag16 f;
#pragma unroll
    for (int r = 0; r < 8; ++r) f.u[r] = p[frag_kpair(r, g) >> 1];
    return f.b;
}

// Same pattern but from an LDS u32 array of packed bf16 pairs, row stride ldu u32s.
__device__ __forceinline__ v16bf load_frag_lds(const unsigned int* p32,
                                               int ldu, int koff, int lane) {
    const int g = lane >> 4, m = lane & 15;
    Frag16 f;
#pragma unroll
    for (int r = 0; r < 8; ++r)
        f.u[r] = p32[m * ldu + ((frag_kpair(r, g) + koff) >> 1)];
    return f.b;
}

__device__ __forceinline__ v8f wmma_bf16(v16bf a, v16bf b, v8f c) {
    return __builtin_amdgcn_wmma_f32_16x16x32_bf16(false, a, false, b,
                                                   (short)0, c, false, false);
}

// ---------------- fp32 -> bf16 convert (x4 vectorized) ----------------
__global__ void cvt_bf16x4(const float4* __restrict__ src,
                           uint2* __restrict__ dst, int n4) {
    int i = blockIdx.x * blockDim.x + threadIdx.x;
    if (i < n4) {
        float4 v = src[i];
        uint2 o;
        o.x = pack2bf(v.x, v.y);
        o.y = pack2bf(v.z, v.w);
        dst[i] = o;
    }
}

// ---------------- GEMM: C = (A @ W^T + bias) * scale ----------------
// A: [M, Kd] bf16 row-major, W: [Nc, Kd] bf16 row-major (torch Linear weight).
// Register-blocked: each wave computes a 32x64 output tile (2 A-frags x 4 B-frags,
// 8 accumulators -> 8 WMMA per 6 fragment loads, ~21 FLOP/B vs 8 unblocked).
// mode 0: C bf16 [M, Nc]. mode 1: per-batch transposed store C[b, col, token] (VO^T).
__global__ void gemm_bt(const unsigned short* __restrict__ A,
                        const unsigned short* __restrict__ W,
                        const float* __restrict__ bias,
                        unsigned short* __restrict__ C,
                        int M, int Nc, int Kd, float scale, int mode) {
    const int lane = threadIdx.x & 31;
    const int wave = threadIdx.x >> 5;
    const int m0 = blockIdx.x * 32;                    // 2 m-frags
    const int n0 = blockIdx.y * 256 + wave * 64;       // 4 n-frags per wave

    v8f acc[2][4] = {};
    for (int ko = 0; ko < Kd; ko += 32) {
        v16bf a0 = load_frag_g(A, m0,      Kd, ko, lane);
        v16bf a1 = load_frag_g(A, m0 + 16, Kd, ko, lane);
        v16bf b0 = load_frag_g(W, n0,      Kd, ko, lane);
        v16bf b1 = load_frag_g(W, n0 + 16, Kd, ko, lane);
        v16bf b2 = load_frag_g(W, n0 + 32, Kd, ko, lane);
        v16bf b3 = load_frag_g(W, n0 + 48, Kd, ko, lane);
        acc[0][0] = wmma_bf16(a0, b0, acc[0][0]);
        acc[0][1] = wmma_bf16(a0, b1, acc[0][1]);
        acc[0][2] = wmma_bf16(a0, b2, acc[0][2]);
        acc[0][3] = wmma_bf16(a0, b3, acc[0][3]);
        acc[1][0] = wmma_bf16(a1, b0, acc[1][0]);
        acc[1][1] = wmma_bf16(a1, b1, acc[1][1]);
        acc[1][2] = wmma_bf16(a1, b2, acc[1][2]);
        acc[1][3] = wmma_bf16(a1, b3, acc[1][3]);
    }

    const int g = lane >> 4, nn = lane & 15;
#pragma unroll
    for (int mi = 0; mi < 2; ++mi) {
#pragma unroll
        for (int ni = 0; ni < 4; ++ni) {
            const int col = n0 + ni * 16 + nn;
            const float bv = bias ? bias[col] : 0.0f;
#pragma unroll
            for (int r = 0; r < 8; ++r) {
                int row = m0 + mi * 16 + r + g * 8;
                float v = (acc[mi][ni][r] + bv) * scale;
                if (mode == 0) {
                    C[(size_t)row * Nc + col] = f2bf(v);
                } else {                   // per-batch transpose: [b, col, token]
                    int bb = row >> 8, tok = row & 255;
                    C[((size_t)bb * Nc + col) * cN + tok] = f2bf(v);
                }
            }
        }
    }
}

// ---------------- fused attention ----------------
// grid (N/16, B), 128 threads (4 waves, 4 heads each).
// Dynamic LDS: attacc[4096] f32 | S[4][4096] f32 | P[4][2048] u32  = 112 KB
// (well under CDNA5's 320 KB/WGP).
__global__ void attn_kernel(const unsigned short* __restrict__ Q,   // scaled, bf16 [B,N,F]
                            const unsigned short* __restrict__ K,   // bf16 [B,N,F]
                            const unsigned short* __restrict__ voT, // bf16 [B,F,N]
                            const int*   __restrict__ adj,          // [B,N,N]
                            const float* __restrict__ lb,           // [B,N,N]
                            const float* __restrict__ bo,           // [F]
                            float* __restrict__ out,                // [B,N,F]
                            float* __restrict__ att) {              // [B,N,N]
    extern __shared__ float smem[];
    float*        attacc = smem;                         // 4096 f32
    float*        Sbase  = smem + 4096;                  // 4 * 4096 f32
    unsigned int* Pbase  = (unsigned int*)(smem + 4096 + 4 * 4096); // 4 * 2048 u32

    const int lane = threadIdx.x & 31;
    const int wave = threadIdx.x >> 5;
    const int q0   = blockIdx.x * 16;
    const int b    = blockIdx.y;

    float*        S = Sbase + wave * 4096;
    unsigned int* P = Pbase + wave * 2048;

    for (int i = threadIdx.x; i < 4096; i += blockDim.x) attacc[i] = 0.0f;
    __syncthreads();

    const unsigned short* Qb = Q   + (size_t)b * cN * cF;
    const unsigned short* Kb = K   + (size_t)b * cN * cF;
    const unsigned short* Vb = voT + (size_t)b * cF * cN;

    const int g = lane >> 4, nn = lane & 15;

    for (int hi = 0; hi < 4; ++hi) {
        const int h   = wave + hi * 4;
        const int dof = h * cHD;

        // ---- scores: S[16,256] = (Q_scaled tile) @ K^T ----
        v16bf qa0 = load_frag_g(Qb, q0, cF, dof,      lane);
        v16bf qa1 = load_frag_g(Qb, q0, cF, dof + 32, lane);
        for (int j = 0; j < 16; ++j) {
            v8f acc = {};
            v16bf kb0 = load_frag_g(Kb, j * 16, cF, dof,      lane);
            v16bf kb1 = load_frag_g(Kb, j * 16, cF, dof + 32, lane);
            acc = wmma_bf16(qa0, kb0, acc);
            acc = wmma_bf16(qa1, kb1, acc);
#pragma unroll
            for (int r = 0; r < 8; ++r)
                S[(r + g * 8) * 256 + j * 16 + nn] = acc[r];
        }
        __syncthreads();

        // ---- mask + bias + softmax over keys; pack P to bf16 ----
        {
            const int r = lane & 15;
            const int half = lane >> 4;
            const int c0 = half * 128;
            float* Srow = S + r * 256 + c0;
            const int*   arow = adj + ((size_t)b * cN + q0 + r) * cN + c0;
            const float* brow = lb  + ((size_t)b * cN + q0 + r) * cN + c0;

            float mx = -INFINITY;
            for (int c = 0; c < 128; ++c) {
                float sp = (arow[c] > 0 ? Srow[c] : NEGV) + brow[c];
                Srow[c] = sp;
                mx = fmaxf(mx, sp);
            }
            mx = fmaxf(mx, __shfl_xor(mx, 16, 32));   // join row halves (wave32)

            float sum = 0.0f;
            for (int c = 0; c < 128; ++c) {
                float e = __expf(Srow[c] - mx);
                Srow[c] = e;
                sum += e;
            }
            sum += __shfl_xor(sum, 16, 32);
            const float inv = 1.0f / sum;

            unsigned int* Prow = P + r * 128 + (c0 >> 1);
            for (int c = 0; c < 128; c += 2) {
                float p0 = Srow[c]     * inv;
                float p1 = Srow[c + 1] * inv;
                Prow[c >> 1] = pack2bf(p0, p1);
                atomicAdd(&attacc[r * 256 + c0 + c],     p0);   // ds_add_f32
                atomicAdd(&attacc[r * 256 + c0 + c + 1], p1);
            }
        }
        __syncthreads();

        // ---- out tile: P[16,256] @ VO^T  (+ bo, since sum_k p == 1) ----
        for (int nt = 0; nt < 4; ++nt) {
            v8f acc = {};
            const int gcol = dof + nt * 16;            // column base in F
            for (int ks = 0; ks < 8; ++ks) {
                v16bf pa = load_frag_lds(P, 128, ks * 32, lane);
                v16bf vb = load_frag_g(Vb, gcol, cN, ks * 32, lane);
                acc = wmma_bf16(pa, vb, acc);
            }
            const int col = gcol + nn;
            const float bb = bo[col];
#pragma unroll
            for (int r = 0; r < 8; ++r) {
                int row = q0 + r + g * 8;
                out[((size_t)b * cN + row) * cF + col] = acc[r] + bb;
            }
        }
        __syncthreads();
    }

    // ---- att_mean = sum_h p / H ----
    for (int i = threadIdx.x; i < 4096; i += blockDim.x) {
        int r = i >> 8, c = i & 255;
        att[((size_t)b * cN + q0 + r) * cN + c] = attacc[i] * (1.0f / (float)cH);
    }
}

// ---------------- host launch ----------------
extern "C" void kernel_launch(void* const* d_in, const int* in_sizes, int n_in,
                              void* d_out, int out_size, void* d_ws, size_t ws_size,
                              hipStream_t stream) {
    const float* obj   = (const float*)d_in[0];
    const float* cross = (const float*)d_in[1];
    const int*   adj   = (const int*)  d_in[2];
    const float* lb    = (const float*)d_in[3];
    const float* Wq    = (const float*)d_in[4];
    const float* bq    = (const float*)d_in[5];
    const float* Wk    = (const float*)d_in[6];
    const float* bk    = (const float*)d_in[7];
    const float* Wv    = (const float*)d_in[8];
    const float* bv    = (const float*)d_in[9];
    const float* Wo    = (const float*)d_in[10];
    const float* bo    = (const float*)d_in[11];

    const size_t ACT = (size_t)cB * cN * cF;   // 4,194,304
    const size_t WEL = (size_t)cF * cF;        // 1,048,576

    unsigned short* obj16   = (unsigned short*)d_ws;
    unsigned short* cross16 = obj16   + ACT;
    unsigned short* wq16    = cross16 + ACT;
    unsigned short* wk16    = wq16    + WEL;
    unsigned short* wv16    = wk16    + WEL;
    unsigned short* wo16    = wv16    + WEL;
    unsigned short* Qb      = wo16    + WEL;
    unsigned short* Kb      = Qb      + ACT;
    unsigned short* Vb      = Kb      + ACT;
    unsigned short* voT     = Vb      + ACT;   // [B, F, N]

    float* out = (float*)d_out;                        // [B,N,F]
    float* att = out + ACT;                            // [B,N,N]

    // 1) fp32 -> bf16 conversions
    {
        int na4 = (int)(ACT / 4), nw4 = (int)(WEL / 4);
        int tb = 256;
        cvt_bf16x4<<<(na4 + tb - 1) / tb, tb, 0, stream>>>((const float4*)obj,   (uint2*)obj16,   na4);
        cvt_bf16x4<<<(na4 + tb - 1) / tb, tb, 0, stream>>>((const float4*)cross, (uint2*)cross16, na4);
        cvt_bf16x4<<<(nw4 + tb - 1) / tb, tb, 0, stream>>>((const float4*)Wq, (uint2*)wq16, nw4);
        cvt_bf16x4<<<(nw4 + tb - 1) / tb, tb, 0, stream>>>((const float4*)Wk, (uint2*)wk16, nw4);
        cvt_bf16x4<<<(nw4 + tb - 1) / tb, tb, 0, stream>>>((const float4*)Wv, (uint2*)wv16, nw4);
        cvt_bf16x4<<<(nw4 + tb - 1) / tb, tb, 0, stream>>>((const float4*)Wo, (uint2*)wo16, nw4);
    }

    // 2) projections (scale 1/sqrt(HD) folded into Q)
    const int M = cB * cN;                 // 4096
    dim3 ggrid(M / 32, cF / 256);          // 128 x 4 blocks, 128 threads (4 waves)
    gemm_bt<<<ggrid, 128, 0, stream>>>(obj16,   wq16, bq,      Qb,  M, cF, cF, 0.125f, 0);
    gemm_bt<<<ggrid, 128, 0, stream>>>(cross16, wk16, bk,      Kb,  M, cF, cF, 1.0f,   0);
    gemm_bt<<<ggrid, 128, 0, stream>>>(cross16, wv16, bv,      Vb,  M, cF, cF, 1.0f,   0);
    // VO = V @ Wo^T, stored transposed per batch as [B, F, N] (bias bo added post-softmax)
    gemm_bt<<<ggrid, 128, 0, stream>>>(Vb,      wo16, nullptr, voT, M, cF, cF, 1.0f,   1);

    // 3) fused attention (scores -> mask/bias -> softmax -> PV -> +bo; head-mean)
    dim3 agrid(cN / 16, cB);
    size_t lds = (size_t)(4096 + 4 * 4096 + 4 * 2048) * 4;   // 112 KB dynamic LDS
    attn_kernel<<<agrid, 128, lds, stream>>>(Qb, Kb, voT, adj, lb, bo, out, att);
}
